// MultiHeadAttention_82368882802725
// MI455X (gfx1250) — matrix operand order repaired and checked by reading, not compiled, not run on hardware
//
#include <hip/hip_runtime.h>
#include <hip/hip_bf16.h>

typedef __attribute__((ext_vector_type(16))) _Float16 v16h;
typedef __attribute__((ext_vector_type(8)))  float    v8f;

#define B_    4
#define S_    2048
#define D_    768
#define H_    12
#define DK_   64
#define MTOT  (B_ * S_)   // 8192

// ---------------------------------------------------------------------------
// Fragment loaders — lane layouts per CDNA5 ISA 7.12.2 (wave32).
// ---------------------------------------------------------------------------

// 16-bit A-matrix 16x32 (MxK), row-major source with leading dim lda.
// lanes 0-15: M=lane, K = {0..7, 16..23}; lanes 16-31: M=lane-16, K = {8..15, 24..31}
__device__ inline v16h load_A16(const _Float16* __restrict__ src, int lda, int lane) {
    const int m  = lane & 15;
    const int kb = (lane >> 4) << 3;              // 0 or 8
    const _Float16* p = src + (size_t)m * lda;
    v16h r;
#pragma unroll
    for (int v = 0; v < 4; ++v) {
        r[2 * v]     = p[kb + 2 * v];
        r[2 * v + 1] = p[kb + 2 * v + 1];
    }
#pragma unroll
    for (int v = 4; v < 8; ++v) {
        r[2 * v]     = p[16 + kb + 2 * (v - 4)];
        r[2 * v + 1] = p[16 + kb + 2 * (v - 4) + 1];
    }
    return r;
}

// 16-bit B-matrix 32x16 (KxN) sourced from TRANSPOSED storage: src is row-major
// [N x K] (B[k][n] = src[n*ld + k]).  lanes 0-15: N=lane, K = {0..7, 16..23};
// lanes 16-31: N=lane-16, K = {8..15, 24..31}; per-lane reads are 2 x b128.
__device__ inline v16h load_BT16(const _Float16* __restrict__ src, int ld, int lane) {
    const int n  = lane & 15;
    const int kb = (lane >> 4) << 3;              // 0 or 8
    const _Float16* p = src + (size_t)n * ld + kb;
    v16h r;
#pragma unroll
    for (int v = 0; v < 8; ++v) {
        const int o = (v < 4) ? 2 * v : 16 + 2 * (v - 4);
        r[2 * v]     = p[o];
        r[2 * v + 1] = p[o + 1];
    }
    return r;
}

__device__ inline v8f wmma_f16(v16h a, v16h b, v8f c) {
    return __builtin_amdgcn_wmma_f32_16x16x32_f16(false, a, false, b, (short)0, c,
                                                  false, false);
}

// Reduce across the 16-lane half-wave (columns of a score tile row).
__device__ inline float redmax16(float x) {
#pragma unroll
    for (int m = 1; m < 16; m <<= 1) x = fmaxf(x, __shfl_xor(x, m, 32));
    return x;
}
__device__ inline float redsum16(float x) {
#pragma unroll
    for (int m = 1; m < 16; m <<= 1) x += __shfl_xor(x, m, 32);
    return x;
}

// Async global -> LDS copy of 16 bytes (CDNA5 ASYNCcnt path; no VGPR data).
__device__ inline void async_copy_b128(const void* gsrc, void* ldst) {
    const unsigned lds_off = (unsigned)(unsigned long long)ldst;  // addr[31:0] = LDS addr
    asm volatile("global_load_async_to_lds_b128 %0, %1, off"
                 :: "v"(lds_off), "v"(gsrc) : "memory");
}

// ---------------------------------------------------------------------------
// fp32 -> fp16 convert
// ---------------------------------------------------------------------------
__global__ void cvt_f32_f16_kernel(const float* __restrict__ src,
                                   _Float16* __restrict__ dst, int n) {
    for (int i = blockIdx.x * blockDim.x + threadIdx.x; i < n;
         i += gridDim.x * blockDim.x)
        dst[i] = (_Float16)src[i];
}

// ---------------------------------------------------------------------------
// GEMM: C = A[M x K](f16) * W[N x K]^T + bias[N].
// One wave -> 16(M) x 64(N) tile; block = 4 waves stacked in M; grid (M/64, N/64).
// HALF_OUT: f16 output.  VT: store head-transposed f16 at [B,H,DK,S]
// (used for the V projection so attention's P*V B-operand is contiguous).
// ---------------------------------------------------------------------------
template <bool HALF_OUT, bool VT>
__global__ __launch_bounds__(128) void gemm_xwT_kernel(
    const _Float16* __restrict__ A, const _Float16* __restrict__ W,
    const float* __restrict__ bias, void* __restrict__ Cout, int M, int N, int K) {
    const int lane = threadIdx.x & 31;
    const int wave = threadIdx.x >> 5;
    const int m0 = blockIdx.x * 64 + wave * 16;
    const int n0 = blockIdx.y * 64;

    v8f acc[4] = {};
    const _Float16* Arow = A + (size_t)m0 * K;

    for (int kk = 0; kk < K; kk += 32) {
        if (kk + 32 < K)  // emits global_prefetch_b8
            __builtin_prefetch(Arow + (lane & 15) * (size_t)K + kk + 32, 0, 0);
        v16h a = load_A16(Arow + kk, K, lane);
#pragma unroll
        for (int j = 0; j < 4; ++j) {
            v16h b = load_BT16(W + (size_t)(n0 + 16 * j) * K + kk, K, lane);
            acc[j] = wmma_f16(a, b, acc[j]);
        }
    }

    const int n  = lane & 15;
    const int mb = (lane >> 4) * 8;
#pragma unroll
    for (int j = 0; j < 4; ++j) {
        const float bv = bias[n0 + 16 * j + n];
        if (HALF_OUT && VT) {
            _Float16* C = (_Float16*)Cout;
            const int ng = n0 + 16 * j + n;
            const int hh = ng >> 6, dk = ng & (DK_ - 1);
#pragma unroll
            for (int v = 0; v < 8; ++v) {
                const int m = m0 + mb + v;
                const int bb = m >> 11, ss = m & (S_ - 1);  // 8 consecutive ss
                C[(((size_t)bb * H_ + hh) * DK_ + dk) * S_ + ss] =
                    (_Float16)(acc[j][v] + bv);
            }
        } else if (HALF_OUT) {
            _Float16* C = (_Float16*)Cout;
#pragma unroll
            for (int v = 0; v < 8; ++v)
                C[(size_t)(m0 + mb + v) * N + n0 + 16 * j + n] =
                    (_Float16)(acc[j][v] + bv);
        } else {
            float* C = (float*)Cout;
#pragma unroll
            for (int v = 0; v < 8; ++v)
                C[(size_t)(m0 + mb + v) * N + n0 + 16 * j + n] = acc[j][v] + bv;
        }
    }
}

// ---------------------------------------------------------------------------
// Online softmax tile update; MASKED selects the causal-edge path.
// ---------------------------------------------------------------------------
template <bool MASKED>
__device__ inline void softmax_update(v8f& s0, v8f& s1, int q0, int mb, int n,
                                      int k0, float* mrow, float* lrow,
                                      float* alpha, _Float16 (*ldsPw)[32]) {
#pragma unroll
    for (int v = 0; v < 8; ++v) {
        float x0 = s0[v] * 0.125f;  // 1/sqrt(64)
        float x1 = s1[v] * 0.125f;
        if (MASKED) {
            const int q = q0 + mb + v;
            x0 = (k0 + n      <= q) ? x0 : -3.0e38f;
            x1 = (k0 + 16 + n <= q) ? x1 : -3.0e38f;
        }
        float mx   = redmax16(fmaxf(x0, x1));
        float mnew = fmaxf(mrow[v], mx);
        alpha[v]   = __expf(mrow[v] - mnew);
        mrow[v]    = mnew;
        float p0 = __expf(x0 - mnew);
        float p1 = __expf(x1 - mnew);
        lrow[v] = lrow[v] * alpha[v] + redsum16(p0 + p1);
        ldsPw[mb + v][n]      = (_Float16)p0;
        ldsPw[mb + v][16 + n] = (_Float16)p1;
    }
}

// Stage one 32-key K tile (32x64) and Vt tile (64x32) into LDS.
// Straight-line: each of the 128 threads issues exactly 2+2 async b128 copies
// (4 async instructions per wave -> ASYNCcnt +4 per chunk per wave).
__device__ inline void stage_tiles(const _Float16* __restrict__ gK,
                                   const _Float16* __restrict__ gV,
                                   _Float16* dK, _Float16* dV, int t) {
    {
        const int r0 = t >> 3, c0 = t & 7;
        async_copy_b128(&gK[(size_t)r0 * D_ + c0 * 8], &dK[r0 * 64 + c0 * 8]);
        const int i = t + 128, r1 = i >> 3, c1 = i & 7;
        async_copy_b128(&gK[(size_t)r1 * D_ + c1 * 8], &dK[r1 * 64 + c1 * 8]);
    }
    {
        const int r0 = t >> 2, c0 = t & 3;
        async_copy_b128(&gV[(size_t)r0 * S_ + c0 * 8], &dV[r0 * 32 + c0 * 8]);
        const int i = t + 128, r1 = i >> 2, c1 = i & 3;
        async_copy_b128(&gV[(size_t)r1 * S_ + c1 * 8], &dV[r1 * 32 + c1 * 8]);
    }
}

// ---------------------------------------------------------------------------
// Causal flash attention.  Q: f16 [B*S, D] (head at col h*64).  Vt: f16
// [B,H,DK,S].  4 waves/block = 4 consecutive q-tiles of one (b,h); each 32-key
// K/V tile is async-staged into double-buffered LDS and shared by all 4 waves.
// ASYNCcnt completes in order, so `s_wait_asynccnt 4` drains chunk c's copies
// while chunk c+1's remain in flight under the WMMAs.  Uniform chunk loop with
// per-wave predicated compute; maskless fast path below the diagonal.
// Grid: (S/64, B*H), block 128.
// ---------------------------------------------------------------------------
__global__ __launch_bounds__(128) void attn_causal_kernel(
    const _Float16* __restrict__ Qh, const _Float16* __restrict__ Kh,
    const _Float16* __restrict__ Vt, _Float16* __restrict__ Oh) {
    __shared__ __align__(16) _Float16 ldsK[2][32 * 64];   // [buf][key][d]
    __shared__ __align__(16) _Float16 ldsV[2][64 * 32];   // [buf][dk][key]
    __shared__ __align__(16) _Float16 ldsP[4][16][32];    // per-wave P tile

    const int t    = threadIdx.x;
    const int lane = t & 31;
    const int wave = t >> 5;
    const int b    = blockIdx.y / H_;
    const int h    = blockIdx.y % H_;
    const int qt0  = blockIdx.x * 4;
    const int qt   = qt0 + wave;
    const int q0   = qt * 16;

    const size_t headbase = (size_t)b * S_ * D_ + (size_t)h * DK_;
    const size_t vtbase   = ((size_t)(b * H_ + h) * DK_) * S_;
    const _Float16* Qp = Qh + headbase + (size_t)q0 * D_;

    // Q A-fragments for d = 0..31 and 32..63 (loaded once per q-tile)
    const v16h qa0 = load_A16(Qp, D_, lane);
    const v16h qa1 = load_A16(Qp + 32, D_, lane);

    v8f o[4] = {};
    float mrow[8], lrow[8];
#pragma unroll
    for (int v = 0; v < 8; ++v) { mrow[v] = -1.0e38f; lrow[v] = 0.0f; }

    const int n  = lane & 15;
    const int mb = (lane >> 4) * 8;

    const int my_chunks  = (qt + 2) >> 1;    // ceil(16*(qt+1)/32)
    const int blk_chunks = (qt0 + 5) >> 1;   // chunk count of the last wave

    // prologue: stage chunk 0 into buffer 0
    stage_tiles(Kh + headbase, Vt + vtbase, ldsK[0], ldsV[0], t);

    for (int c = 0; c < blk_chunks; ++c) {
        const int buf = c & 1;

        if (c + 1 < blk_chunks) {
            // issue next chunk into the other buffer, then drain chunk c only
            const int k1 = (c + 1) * 32;
            stage_tiles(Kh + headbase + (size_t)k1 * D_, Vt + vtbase + k1,
                        ldsK[buf ^ 1], ldsV[buf ^ 1], t);
            asm volatile("s_wait_asynccnt 0x4" ::: "memory");
        } else {
            asm volatile("s_wait_asynccnt 0x0" ::: "memory");
        }
        __syncthreads();  // all waves' chunk-c copies visible block-wide

        if (c < my_chunks) {
            const int k0 = c * 32;
            const _Float16* lK = ldsK[buf];
            const _Float16* lV = ldsV[buf];

            // scores: two 16x16 tiles (keys k0..k0+15, k0+16..k0+31), from LDS
            v8f s0 = {}, s1 = {};
            s0 = wmma_f16(qa0, load_BT16(lK, 64, lane), s0);
            s0 = wmma_f16(qa1, load_BT16(lK + 32, 64, lane), s0);
            s1 = wmma_f16(qa0, load_BT16(lK + 16 * 64, 64, lane), s1);
            s1 = wmma_f16(qa1, load_BT16(lK + 16 * 64 + 32, 64, lane), s1);

            float alpha[8];
            if (k0 + 31 <= q0)  // fully below diagonal: no mask needed
                softmax_update<false>(s0, s1, q0, mb, n, k0, mrow, lrow, alpha,
                                      ldsP[wave]);
            else
                softmax_update<true>(s0, s1, q0, mb, n, k0, mrow, lrow, alpha,
                                     ldsP[wave]);

            asm volatile("s_wait_dscnt 0" ::: "memory");  // wave-local LDS RAW
            const v16h pa = load_A16(&ldsP[wave][0][0], 32, lane);

#pragma unroll
            for (int j = 0; j < 4; ++j)
#pragma unroll
                for (int v = 0; v < 8; ++v) o[j][v] *= alpha[v];

            // o += P(16x32) * V(32x64); Vt tile gives contiguous B-fragments
#pragma unroll
            for (int j = 0; j < 4; ++j)
                o[j] = wmma_f16(pa, load_BT16(lV + 16 * j * 32, 32, lane), o[j]);
        }
        __syncthreads();  // buffer `buf` free before it is re-staged (c+2)
    }

    // normalize + store f16 at [b, q, h*64 + dk]
#pragma unroll
    for (int j = 0; j < 4; ++j)
#pragma unroll
        for (int v = 0; v < 8; ++v)
            Oh[headbase + (size_t)(q0 + mb + v) * D_ + 16 * j + n] =
                (_Float16)(o[j][v] / lrow[v]);
}

// ---------------------------------------------------------------------------
extern "C" void kernel_launch(void* const* d_in, const int* in_sizes, int n_in,
                              void* d_out, int out_size, void* d_ws, size_t ws_size,
                              hipStream_t stream) {
    const float* x  = (const float*)d_in[0];
    const float* wq = (const float*)d_in[1];
    const float* bq = (const float*)d_in[2];
    const float* wk = (const float*)d_in[3];
    const float* bk = (const float*)d_in[4];
    const float* wv = (const float*)d_in[5];
    const float* bv = (const float*)d_in[6];
    const float* wo = (const float*)d_in[7];
    const float* bo = (const float*)d_in[8];

    const size_t act_sz = (size_t)MTOT * D_ * sizeof(_Float16);  // 12.6 MB
    const size_t w_sz   = (size_t)D_ * D_ * sizeof(_Float16);    // 1.18 MB

    char* ws = (char*)d_ws;
    _Float16* xh  = (_Float16*)ws; ws += act_sz;  // later reused for attn out
    _Float16* qh  = (_Float16*)ws; ws += act_sz;
    _Float16* kh  = (_Float16*)ws; ws += act_sz;
    _Float16* vth = (_Float16*)ws; ws += act_sz;  // V stored [B,H,DK,S]
    _Float16* wqh = (_Float16*)ws; ws += w_sz;
    _Float16* wkh = (_Float16*)ws; ws += w_sz;
    _Float16* wvh = (_Float16*)ws; ws += w_sz;
    _Float16* woh = (_Float16*)ws; ws += w_sz;
    _Float16* ah  = xh;  // x no longer needed after QKV projections

    // fp32 -> fp16 conversions
    {
        const int nx = MTOT * D_;
        const int nw = D_ * D_;
        cvt_f32_f16_kernel<<<2048, 256, 0, stream>>>(x, xh, nx);
        cvt_f32_f16_kernel<<<1024, 256, 0, stream>>>(wq, wqh, nw);
        cvt_f32_f16_kernel<<<1024, 256, 0, stream>>>(wk, wkh, nw);
        cvt_f32_f16_kernel<<<1024, 256, 0, stream>>>(wv, wvh, nw);
        cvt_f32_f16_kernel<<<1024, 256, 0, stream>>>(wo, woh, nw);
    }

    // QKV projections (f16 out); V goes out head-transposed
    dim3 ggrid(MTOT / 64, D_ / 64);
    gemm_xwT_kernel<true, false><<<ggrid, 128, 0, stream>>>(xh, wqh, bq, qh,
                                                            MTOT, D_, D_);
    gemm_xwT_kernel<true, false><<<ggrid, 128, 0, stream>>>(xh, wkh, bk, kh,
                                                            MTOT, D_, D_);
    gemm_xwT_kernel<true, true><<<ggrid, 128, 0, stream>>>(xh, wvh, bv, vth,
                                                           MTOT, D_, D_);

    // causal attention -> ah (f16, [B*S, D] head-interleaved layout)
    dim3 agrid(S_ / 64, B_ * H_);
    attn_causal_kernel<<<agrid, 128, 0, stream>>>(qh, kh, vth, ah);

    // output projection: d_out(f32) = ah * wo^T + bo
    gemm_xwT_kernel<false, false><<<ggrid, 128, 0, stream>>>(ah, woh, bo, d_out,
                                                             MTOT, D_, D_);
}
